// Encoder_11364483465632
// MI455X (gfx1250) — compile-verified, matrix operand order checked by reading, not run on hardware
//
#include <hip/hip_runtime.h>

// ---------------- types for CDNA5 WMMA (wave32) ----------------
typedef __attribute__((ext_vector_type(16))) _Float16 v16h;
typedef __attribute__((ext_vector_type(8)))  _Float16 h8;
typedef __attribute__((ext_vector_type(8)))  float    v8f;

#define B_  4
#define S_  2048
#define F_  512
#define D_  512
#define H_  8
#define DH_ 64
#define DF_ 2048
#define L_  4

__device__ __forceinline__ v8f wmma_f16(v16h a, v16h b, v8f c) {
  // D = A(16x32 f16) x B(32x16 f16) + C(16x16 f32)
  return __builtin_amdgcn_wmma_f32_16x16x32_f16(false, a, false, b,
                                                (short)0, c, false, false);
}

// Load one 16x32 f16 fragment row (A layout) / col (B layout) for this lane.
// Element e -> k = (e/8)*16 + g*8 + (e%8), so two contiguous 16B chunks.
__device__ __forceinline__ v16h ld_frag32(const _Float16* row, int g) {
  h8 lo = *(const h8*)(row + g * 8);
  h8 hi = *(const h8*)(row + 16 + g * 8);
  return __builtin_shufflevector(lo, hi, 0,1,2,3,4,5,6,7,8,9,10,11,12,13,14,15);
}

// ---------------- positional encoding table ----------------
__global__ void pe_kernel(float* __restrict__ pe) {
  int idx = blockIdx.x * 256 + threadIdx.x;
  if (idx >= S_ * D_) return;
  int s = idx / D_, d = idx % D_;
  float expo = (float)(d & ~1) / (float)D_;
  float angle = (float)s * __powf(10000.0f, -expo);
  pe[idx] = (d & 1) ? __cosf(angle) : __sinf(angle);
}

// ---------------- weight packing (fp32 -> fp16, transposed) ----------------
// dst[n*K + k] = src[k*N + n]
__global__ void pack_transpose(const float* __restrict__ src,
                               _Float16* __restrict__ dst, int K, int N) {
  int idx = blockIdx.x * 256 + threadIdx.x;
  if (idx >= K * N) return;
  int n = idx / K, k = idx % K;
  dst[idx] = (_Float16)src[(size_t)k * N + n];
}

// Wq[l] is [H][D][DH]; fold heads into cols: dst[n*D + d] = src[(n/64)][d][n%64]
__global__ void pack_qkv(const float* __restrict__ src,
                         _Float16* __restrict__ dst) {
  int idx = blockIdx.x * 256 + threadIdx.x;
  if (idx >= D_ * D_) return;
  int n = idx / D_, d = idx % D_;
  int hh = n >> 6, e = n & 63;
  dst[idx] = (_Float16)src[((size_t)hh * D_ + d) * DH_ + e];
}

// ---------------- tiled WMMA GEMM ----------------
// out[M,N] = A[M,K] * Wt^T  (Wt stored [N,K] fp16, row-major) + bias epilogue
// EPI: 0 = +bias, 1 = +bias,relu, 2 = +bias + pe[row%S]
// 128 threads = 4 waves, each wave owns a 64x64 output tile (16 WMMAs/chunk).
template <typename TA, typename TO, int EPI>
__global__ __launch_bounds__(128)
void gemm_wmma(const TA* __restrict__ A, const _Float16* __restrict__ Wt,
               const float* __restrict__ bias, const float* __restrict__ pe,
               TO* __restrict__ out, int M, int N, int K) {
  __shared__ _Float16 As[128][40];   // 128 rows x 32 K (pad 8)
  __shared__ _Float16 Bs[128][40];   // 128 cols x 32 K (pad 8)
  const int t    = threadIdx.x;      // 0..127, one A-row / one W-col each
  const int lane = t & 31, lr = lane & 15, g = (lane >> 4) & 1;
  const int w = t >> 5, wm = w >> 1, wn = w & 1;   // wave grid 2x2
  const int rowBase = blockIdx.y * 128;
  const int colBase = blockIdx.x * 128;

  const v8f vz = {0.f,0.f,0.f,0.f,0.f,0.f,0.f,0.f};
  v8f acc[4][4];
#pragma unroll
  for (int mt = 0; mt < 4; mt++)
#pragma unroll
    for (int nt = 0; nt < 4; nt++) acc[mt][nt] = vz;

  // register staging for software-pipelined global->LDS
  float4 aF[8];
  h8     aH[4];
  h8     wR[4];

  auto gload = [&](int kc) {
    const char* ap = (const char*)A +
        sizeof(TA) * ((size_t)(rowBase + t) * K + kc);
    if (sizeof(TA) == 4) {
      const float4* p = (const float4*)ap;
#pragma unroll
      for (int i = 0; i < 8; i++) aF[i] = p[i];
    } else {
      const h8* p = (const h8*)ap;
#pragma unroll
      for (int i = 0; i < 4; i++) aH[i] = p[i];
    }
    const h8* q = (const h8*)(Wt + (size_t)(colBase + t) * K + kc);
#pragma unroll
    for (int i = 0; i < 4; i++) wR[i] = q[i];
    if (kc + 32 < K)
      __builtin_prefetch((const void*)(ap + sizeof(TA) * 32), 0, 1);
  };

  auto sstore = [&]() {
    _Float16 tmp[32];
    if (sizeof(TA) == 4) {
#pragma unroll
      for (int i = 0; i < 8; i++) {
        tmp[4*i+0] = (_Float16)aF[i].x; tmp[4*i+1] = (_Float16)aF[i].y;
        tmp[4*i+2] = (_Float16)aF[i].z; tmp[4*i+3] = (_Float16)aF[i].w;
      }
    } else {
#pragma unroll
      for (int i = 0; i < 4; i++)
#pragma unroll
        for (int j = 0; j < 8; j++) tmp[8*i+j] = aH[i][j];
    }
#pragma unroll
    for (int i = 0; i < 4; i++) {
      h8 hv;
#pragma unroll
      for (int j = 0; j < 8; j++) hv[j] = tmp[8*i+j];
      *(h8*)&As[t][8*i] = hv;
      *(h8*)&Bs[t][8*i] = wR[i];
    }
  };

  gload(0);
  for (int kc = 0; kc < K; kc += 32) {
    __syncthreads();            // previous chunk fully consumed
    sstore();
    __syncthreads();            // LDS ready
    if (kc + 32 < K) gload(kc + 32);   // next chunk in flight during WMMAs

    v16h af[4], bf[4];
#pragma unroll
    for (int mt = 0; mt < 4; mt++)
      af[mt] = ld_frag32(&As[wm * 64 + mt * 16 + lr][0], g);
#pragma unroll
    for (int nt = 0; nt < 4; nt++)
      bf[nt] = ld_frag32(&Bs[wn * 64 + nt * 16 + lr][0], g);
#pragma unroll
    for (int mt = 0; mt < 4; mt++)
#pragma unroll
      for (int nt = 0; nt < 4; nt++)
        acc[mt][nt] = wmma_f16(af[mt], bf[nt], acc[mt][nt]);
  }

  // epilogue: C layout row = r + 8*g, col = lr within tile
#pragma unroll
  for (int mt = 0; mt < 4; mt++)
#pragma unroll
    for (int nt = 0; nt < 4; nt++) {
      int colg = colBase + wn * 64 + nt * 16 + lr;
      float bb = bias[colg];
#pragma unroll
      for (int r = 0; r < 8; r++) {
        int row = rowBase + wm * 64 + mt * 16 + g * 8 + r;
        float v = acc[mt][nt][r] + bb;
        if (EPI == 2) v += pe[(size_t)(row & (S_ - 1)) * D_ + colg];
        if (EPI == 1) v = fmaxf(v, 0.0f);
        out[(size_t)row * N + colg] = (TO)v;
      }
    }
}

// ---------------- flash attention (per b,head; 64 query rows / block) -------
__global__ __launch_bounds__(128)
void flash_attn(const _Float16* __restrict__ q, const _Float16* __restrict__ k,
                const _Float16* __restrict__ v, _Float16* __restrict__ heads) {
  __shared__ _Float16 Ks[64][72];
  __shared__ _Float16 Vs[64][72];
  __shared__ _Float16 Ps[4][16][72];
  const int t    = threadIdx.x;
  const int lane = t & 31, lr = lane & 15, g = (lane >> 4) & 1;
  const int w  = t >> 5;                 // wave 0..3
  const int qb = blockIdx.x;             // query block (64 rows)
  const int bh = blockIdx.y;
  const int b = bh / H_, hd = bh % H_;
  const size_t headOff = (size_t)hd * DH_;
  const size_t bOff    = (size_t)b * S_ * D_;
  const int qrow = qb * 64 + w * 16 + lr;

  v16h aq[2];
  {
    const _Float16* qp = q + bOff + (size_t)qrow * D_ + headOff;
    aq[0] = ld_frag32(qp, g);
    aq[1] = ld_frag32(qp + 32, g);
  }

  const v8f vz = {0.f,0.f,0.f,0.f,0.f,0.f,0.f,0.f};
  float mrow[8], lsum[8];
  v8f o_acc[4];
#pragma unroll
  for (int r = 0; r < 8; r++) { mrow[r] = -1e30f; lsum[r] = 0.0f; }
#pragma unroll
  for (int ct = 0; ct < 4; ct++) o_acc[ct] = vz;

  const int key0 = t >> 1;
  const int seg  = (t & 1) << 5;  // 0 / 32

  for (int kb = 0; kb < S_ / 64; kb++) {
    __syncthreads();
    {
      // gfx1250 async copy: global -> LDS directly, tracked by ASYNCcnt.
      // IOFFSET is added to both the LDS and global addresses.
      unsigned kl = (unsigned)(uintptr_t)&Ks[key0][seg];
      unsigned vl = (unsigned)(uintptr_t)&Vs[key0][seg];
      unsigned long long kg = (unsigned long long)(uintptr_t)
          (k + bOff + (size_t)(kb * 64 + key0) * D_ + headOff + seg);
      unsigned long long vg = (unsigned long long)(uintptr_t)
          (v + bOff + (size_t)(kb * 64 + key0) * D_ + headOff + seg);
#pragma unroll
      for (int i = 0; i < 4; i++) {
        asm volatile("global_load_async_to_lds_b128 %0, %1, off offset:%2"
                     :: "v"(kl), "v"(kg), "i"(i * 16) : "memory");
        asm volatile("global_load_async_to_lds_b128 %0, %1, off offset:%2"
                     :: "v"(vl), "v"(vg), "i"(i * 16) : "memory");
      }
      asm volatile("s_wait_asynccnt 0x0" ::: "memory");
    }
    __syncthreads();

    // S = Q * K^T, 16 rows x 64 keys held in 4 C tiles
    v8f s[4];
#pragma unroll
    for (int nt = 0; nt < 4; nt++) {
      v8f a = vz;
      a = wmma_f16(aq[0], ld_frag32(&Ks[nt * 16 + lr][0],  g), a);
      a = wmma_f16(aq[1], ld_frag32(&Ks[nt * 16 + lr][32], g), a);
      s[nt] = a;
    }

    // online softmax: row M = r + 8*g, spread over 16 lanes (one key col each)
#pragma unroll
    for (int r = 0; r < 8; r++) {
      float s0 = s[0][r] * 0.125f, s1 = s[1][r] * 0.125f;
      float s2 = s[2][r] * 0.125f, s3 = s[3][r] * 0.125f;
      float mx = fmaxf(fmaxf(s0, s1), fmaxf(s2, s3));
#pragma unroll
      for (int off = 1; off < 16; off <<= 1) mx = fmaxf(mx, __shfl_xor(mx, off, 32));
      float mnew = fmaxf(mrow[r], mx);
      float p0 = __expf(s0 - mnew), p1 = __expf(s1 - mnew);
      float p2 = __expf(s2 - mnew), p3 = __expf(s3 - mnew);
      float rs = p0 + p1 + p2 + p3;
#pragma unroll
      for (int off = 1; off < 16; off <<= 1) rs += __shfl_xor(rs, off, 32);
      float alpha = __expf(mrow[r] - mnew);
      lsum[r] = lsum[r] * alpha + rs;
      mrow[r] = mnew;
#pragma unroll
      for (int ct = 0; ct < 4; ct++) o_acc[ct][r] *= alpha;
      s[0][r] = p0; s[1][r] = p1; s[2][r] = p2; s[3][r] = p3;
    }

    // transpose P through LDS into A-fragment layout
#pragma unroll
    for (int nt = 0; nt < 4; nt++)
#pragma unroll
      for (int r = 0; r < 8; r++)
        Ps[w][g * 8 + r][nt * 16 + lr] = (_Float16)s[nt][r];
    __syncthreads();

    v16h ap0 = ld_frag32(&Ps[w][lr][0],  g);
    v16h ap1 = ld_frag32(&Ps[w][lr][32], g);
#pragma unroll
    for (int ct = 0; ct < 4; ct++) {
      v16h bv;
#pragma unroll
      for (int e = 0; e < 16; e++) {
        int kk = ((e >> 3) << 4) + g * 8 + (e & 7);
        bv[e] = Vs[kk][ct * 16 + lr];
      }
      o_acc[ct] = wmma_f16(ap0, bv, o_acc[ct]);
#pragma unroll
      for (int e = 0; e < 16; e++) {
        int kk = 32 + ((e >> 3) << 4) + g * 8 + (e & 7);
        bv[e] = Vs[kk][ct * 16 + lr];
      }
      o_acc[ct] = wmma_f16(ap1, bv, o_acc[ct]);
    }
  }

#pragma unroll
  for (int r = 0; r < 8; r++) {
    float inv = 1.0f / lsum[r];
    int row = qb * 64 + w * 16 + g * 8 + r;
    _Float16* op = heads + bOff + (size_t)row * D_ + headOff;
#pragma unroll
    for (int ct = 0; ct < 4; ct++)
      op[ct * 16 + lr] = (_Float16)(o_acc[ct][r] * inv);
  }
}

// ---------------- residual + layernorm (in-place into base) ----------------
__global__ __launch_bounds__(128)
void ln_kernel(float* __restrict__ base, const float* __restrict__ delta,
               const float* __restrict__ gamma, const float* __restrict__ beta) {
  __shared__ float red[8];
  const int row = blockIdx.x;
  const int t = threadIdx.x, lane = t & 31, wv = t >> 5;
  float* bp = base + (size_t)row * D_;
  const float* dp = delta + (size_t)row * D_;
  float x[4], s = 0.f, sq = 0.f;
#pragma unroll
  for (int i = 0; i < 4; i++) {
    int c = t + i * 128;
    x[i] = bp[c] + dp[c];
    s += x[i]; sq += x[i] * x[i];
  }
#pragma unroll
  for (int off = 16; off > 0; off >>= 1) {
    s  += __shfl_xor(s,  off, 32);
    sq += __shfl_xor(sq, off, 32);
  }
  if (lane == 0) { red[wv] = s; red[4 + wv] = sq; }
  __syncthreads();
  s  = red[0] + red[1] + red[2] + red[3];
  sq = red[4] + red[5] + red[6] + red[7];
  float mean = s * (1.0f / D_);
  float var  = sq * (1.0f / D_) - mean * mean;
  float rstd = rsqrtf(var + 1e-6f);
#pragma unroll
  for (int i = 0; i < 4; i++) {
    int c = t + i * 128;
    bp[c] = gamma[c] * (x[i] - mean) * rstd + beta[c];
  }
}

// ---------------- host driver ----------------
extern "C" void kernel_launch(void* const* d_in, const int* in_sizes, int n_in,
                              void* d_out, int out_size, void* d_ws, size_t ws_size,
                              hipStream_t stream) {
  (void)in_sizes; (void)n_in; (void)out_size; (void)ws_size;
  const float* x     = (const float*)d_in[0];
  const float* W_emb = (const float*)d_in[1];
  const float* b_emb = (const float*)d_in[2];
  const float* Wq    = (const float*)d_in[3];
  const float* bq    = (const float*)d_in[4];
  const float* Wk    = (const float*)d_in[5];
  const float* bk    = (const float*)d_in[6];
  const float* Wv    = (const float*)d_in[7];
  const float* bv    = (const float*)d_in[8];
  const float* Wo    = (const float*)d_in[9];
  const float* bo    = (const float*)d_in[10];
  const float* W1    = (const float*)d_in[11];
  const float* b1    = (const float*)d_in[12];
  const float* W2    = (const float*)d_in[13];
  const float* b2    = (const float*)d_in[14];
  const float* g1    = (const float*)d_in[15];
  const float* be1   = (const float*)d_in[16];
  const float* g2    = (const float*)d_in[17];
  const float* be2   = (const float*)d_in[18];

  const size_t M = (size_t)B_ * S_;     // 8192 token rows
  char* wsp = (char*)d_ws;
  size_t off = 0;
  auto alloc = [&](size_t bytes) -> char* {
    char* p = wsp + off;
    off += (bytes + 255) & ~(size_t)255;
    return p;
  };
  float*     pe    = (float*)alloc((size_t)S_ * D_ * 4);
  float*     h     = (float*)alloc(M * D_ * 4);
  float*     t0    = (float*)alloc(M * D_ * 4);
  _Float16*  heads = (_Float16*)alloc(M * D_ * 2);
  _Float16*  qbuf  = (_Float16*)alloc(M * D_ * 2);
  _Float16*  kbuf  = (_Float16*)alloc(M * D_ * 2);
  _Float16*  vbuf  = (_Float16*)alloc(M * D_ * 2);
  _Float16*  fmid  = (_Float16*)alloc(M * DF_ * 2);
  _Float16*  wembT = (_Float16*)alloc((size_t)D_ * F_ * 2);
  _Float16 *wqT[L_], *wkT[L_], *wvT[L_], *woT[L_], *w1T[L_], *w2T[L_];
  for (int l = 0; l < L_; l++) {
    wqT[l] = (_Float16*)alloc((size_t)D_ * D_ * 2);
    wkT[l] = (_Float16*)alloc((size_t)D_ * D_ * 2);
    wvT[l] = (_Float16*)alloc((size_t)D_ * D_ * 2);
    woT[l] = (_Float16*)alloc((size_t)D_ * D_ * 2);
    w1T[l] = (_Float16*)alloc((size_t)DF_ * D_ * 2);
    w2T[l] = (_Float16*)alloc((size_t)D_ * DF_ * 2);
  }

  // --- precompute PE + pack weights to fp16 ---
  pe_kernel<<<(S_ * D_ + 255) / 256, 256, 0, stream>>>(pe);
  pack_transpose<<<(F_ * D_ + 255) / 256, 256, 0, stream>>>(W_emb, wembT, F_, D_);
  for (int l = 0; l < L_; l++) {
    const size_t qo = (size_t)l * H_ * D_ * DH_;
    pack_qkv<<<(D_ * D_ + 255) / 256, 256, 0, stream>>>(Wq + qo, wqT[l]);
    pack_qkv<<<(D_ * D_ + 255) / 256, 256, 0, stream>>>(Wk + qo, wkT[l]);
    pack_qkv<<<(D_ * D_ + 255) / 256, 256, 0, stream>>>(Wv + qo, wvT[l]);
    pack_transpose<<<(D_ * D_ + 255) / 256, 256, 0, stream>>>(
        Wo + (size_t)l * D_ * D_, woT[l], D_, D_);
    pack_transpose<<<(D_ * DF_ + 255) / 256, 256, 0, stream>>>(
        W1 + (size_t)l * D_ * DF_, w1T[l], D_, DF_);
    pack_transpose<<<(DF_ * D_ + 255) / 256, 256, 0, stream>>>(
        W2 + (size_t)l * DF_ * D_, w2T[l], DF_, D_);
  }

  const dim3 blk(128);
  const dim3 gN512(D_ / 128, (int)(M / 128));
  const dim3 gN2048(DF_ / 128, (int)(M / 128));

  // --- embedding + PE ---
  gemm_wmma<float, float, 2><<<gN512, blk, 0, stream>>>(
      x, wembT, b_emb, pe, h, (int)M, D_, F_);

  for (int l = 0; l < L_; l++) {
    // QKV projections (fp16 outputs for attention)
    gemm_wmma<float, _Float16, 0><<<gN512, blk, 0, stream>>>(
        h, wqT[l], bq + (size_t)l * D_, nullptr, qbuf, (int)M, D_, D_);
    gemm_wmma<float, _Float16, 0><<<gN512, blk, 0, stream>>>(
        h, wkT[l], bk + (size_t)l * D_, nullptr, kbuf, (int)M, D_, D_);
    gemm_wmma<float, _Float16, 0><<<gN512, blk, 0, stream>>>(
        h, wvT[l], bv + (size_t)l * D_, nullptr, vbuf, (int)M, D_, D_);
    // attention
    flash_attn<<<dim3(S_ / 64, B_ * H_), dim3(128), 0, stream>>>(
        qbuf, kbuf, vbuf, heads);
    // output projection + residual + LN1
    gemm_wmma<_Float16, float, 0><<<gN512, blk, 0, stream>>>(
        heads, woT[l], bo + (size_t)l * D_, nullptr, t0, (int)M, D_, D_);
    ln_kernel<<<(int)M, 128, 0, stream>>>(
        h, t0, g1 + (size_t)l * D_, be1 + (size_t)l * D_);
    // FFN
    gemm_wmma<float, _Float16, 1><<<gN2048, blk, 0, stream>>>(
        h, w1T[l], b1 + (size_t)l * DF_, nullptr, fmid, (int)M, DF_, D_);
    gemm_wmma<_Float16, float, 0><<<gN512, blk, 0, stream>>>(
        fmid, w2T[l], b2 + (size_t)l * D_, nullptr, t0, (int)M, D_, DF_);
    ln_kernel<<<(int)M, 128, 0, stream>>>(
        h, t0, g2 + (size_t)l * D_, be2 + (size_t)l * D_);
  }

  hipMemcpyAsync(d_out, h, M * D_ * sizeof(float),
                 hipMemcpyDeviceToDevice, stream);
}
